// DoubleStreamBlock_13932873909029
// MI455X (gfx1250) — matrix operand order repaired and checked by reading, not compiled, not run on hardware
//
#include <hip/hip_runtime.h>
#include <math.h>

typedef _Float16 f16;
typedef float v8f __attribute__((ext_vector_type(8)));
typedef _Float16 v16h __attribute__((ext_vector_type(16)));
typedef unsigned int u32x4 __attribute__((ext_vector_type(4)));

union Frag {
    v16h h;
    struct { u32x4 lo, hi; } u;
};

#define D_MODEL 3072
#define L_TOT   1280
#define L_TXT   256
#define N_QKV   9216
#define N_MLP   12288
#define N_MOD   18432
#define EPSV    1e-6f

#if __has_builtin(__builtin_amdgcn_global_load_async_to_lds_b128) && \
    __has_builtin(__builtin_amdgcn_s_wait_asynccnt)
#define USE_ASYNC_LDS 1
// builtin expects: (int __vector_size(16) AS1* src, int __vector_size(16) AS3* dst, Imm off, Imm cpol)
typedef int i32v4 __attribute__((vector_size(16)));
typedef __attribute__((address_space(1))) i32v4 gas_i32v4;
typedef __attribute__((address_space(3))) i32v4 las_i32v4;
#endif

// ---------------------------------------------------------------- silu(vec)
__global__ void silu_kernel(const float* __restrict__ vec, float* __restrict__ sv) {
    int i = blockIdx.x * 256 + threadIdx.x;
    if (i < D_MODEL) { float x = vec[i]; sv[i] = x / (1.f + expf(-x)); }
}

// ------------------------------------------------- mod = silu(vec)@W + b (GEMV)
__global__ __launch_bounds__(256) void mod_gemv_kernel(const float* __restrict__ sv,
                                                       const float* __restrict__ W,
                                                       const float* __restrict__ b,
                                                       float* __restrict__ out) {
    int j = blockIdx.x * 256 + threadIdx.x;          // 0..18431
    float acc = b[j];
    const float* w = W + j;
#pragma unroll 4
    for (int i = 0; i < D_MODEL; ++i) acc += sv[i] * w[(size_t)i * N_MOD];
    out[j] = acc;
}

// --------------------------------- weights: f32 [K][N] -> f16 [N][K] (tiled)
__global__ __launch_bounds__(256) void transpose_convert_kernel(const float* __restrict__ in,
                                                                f16* __restrict__ out,
                                                                int K, int N) {
    __shared__ float tile[32][33];
    int k0 = blockIdx.y * 32, n0 = blockIdx.x * 32;
    int r = threadIdx.x >> 5, c = threadIdx.x & 31;     // 8 x 32
#pragma unroll
    for (int i = 0; i < 4; ++i)
        tile[r + i * 8][c] = in[(size_t)(k0 + r + i * 8) * N + n0 + c];
    __syncthreads();
#pragma unroll
    for (int i = 0; i < 4; ++i)
        out[(size_t)(n0 + r + i * 8) * K + k0 + c] = (f16)tile[c][r + i * 8];
}

// --------------------------- x_mod = (1+scale)*LayerNorm(x)+shift, f16 output
// rows 0..255 come from txtX, rows 256..1279 from imgX (concat order)
__global__ __launch_bounds__(256) void ln_mod_kernel(const float* __restrict__ txtX,
                                                     const float* __restrict__ imgX,
                                                     const float* __restrict__ modT,
                                                     const float* __restrict__ modI,
                                                     f16* __restrict__ out,
                                                     int shOff, int scOff) {
    int row = blockIdx.x;
    const float* src; const float* mod;
    if (row < L_TXT) { src = txtX + (size_t)row * D_MODEL;            mod = modT; }
    else             { src = imgX + (size_t)(row - L_TXT) * D_MODEL;  mod = modI; }
    int tid = threadIdx.x;
    float x[12];
    float s = 0.f;
#pragma unroll
    for (int i = 0; i < 12; ++i) { x[i] = src[tid + i * 256]; s += x[i]; }
    __shared__ float red[8];
#pragma unroll
    for (int m = 16; m > 0; m >>= 1) s += __shfl_xor(s, m, 32);
    if ((tid & 31) == 0) red[tid >> 5] = s;
    __syncthreads();
    float tot = 0.f;
#pragma unroll
    for (int wv = 0; wv < 8; ++wv) tot += red[wv];
    float mean = tot * (1.f / (float)D_MODEL);
    __syncthreads();
    float vs = 0.f;
#pragma unroll
    for (int i = 0; i < 12; ++i) { float d = x[i] - mean; vs += d * d; }
#pragma unroll
    for (int m = 16; m > 0; m >>= 1) vs += __shfl_xor(vs, m, 32);
    if ((tid & 31) == 0) red[tid >> 5] = vs;
    __syncthreads();
    float vtot = 0.f;
#pragma unroll
    for (int wv = 0; wv < 8; ++wv) vtot += red[wv];
    float inv = rsqrtf(vtot * (1.f / (float)D_MODEL) + EPSV);
    const float* sh = mod + shOff;
    const float* sc = mod + scOff;
#pragma unroll
    for (int i = 0; i < 12; ++i) {
        int c = tid + i * 256;
        float y = (x[i] - mean) * inv;
        out[(size_t)row * D_MODEL + c] = (f16)(y * (1.f + sc[c]) + sh[c]);
    }
}

// ------------------------------------------------------------- WMMA GEMM f16
// C[M,N] = A[M,K] @ B[K,N] + bias, with B pre-transposed: Bt[N][K] (ldb == K).
// MODE 0: f32 store.  MODE 1: GELU -> f16.
// Block tile 128x128, K-step 32, 8 waves (2x4), each wave 64x32 (8 C tiles).
// A tile double-buffered in LDS via async global->LDS copies (ASYNCcnt);
// B fragments stream straight from global (two b128 per lane per frag).
template <int MODE>
__global__ __launch_bounds__(256) void gemm_kernel(const f16* __restrict__ A, long lda,
                                                   const f16* __restrict__ Bt, long ldb,
                                                   const float* __restrict__ bias,
                                                   void* __restrict__ Cout, long ldc, int K) {
    __shared__ __align__(16) f16 As[2][128 * 32];   // [row][k], 8KB per buffer
    int tid = threadIdx.x, wave = tid >> 5, lane = tid & 31;
    int l16 = lane & 15, hb = lane >> 4;
    int wm = wave & 1, wn = wave >> 1;
    long mrow0 = (long)blockIdx.y * 128;
    long ncol0 = (long)blockIdx.x * 128;

    v8f acc[4][2];
#pragma unroll
    for (int i = 0; i < 4; ++i)
#pragma unroll
        for (int j = 0; j < 2; ++j)
#pragma unroll
            for (int r = 0; r < 8; ++r) acc[i][j][r] = 0.f;

    // per-thread A-copy coordinates (two b128 chunks: 128x32 f16 = 512 chunks)
    int ar0 = tid >> 2,          ac0 = (tid & 3) * 8;
    int ar1 = (tid + 256) >> 2,  ac1 = ((tid + 256) & 3) * 8;

    auto issueA = [&](int buf, int k0) {
        const f16* s0 = A + (mrow0 + ar0) * lda + k0 + ac0;
        const f16* s1 = A + (mrow0 + ar1) * lda + k0 + ac1;
        f16* d0 = &As[buf][ar0 * 32 + ac0];
        f16* d1 = &As[buf][ar1 * 32 + ac1];
#ifdef USE_ASYNC_LDS
        __builtin_amdgcn_global_load_async_to_lds_b128((gas_i32v4*)s0, (las_i32v4*)d0, 0, 0);
        __builtin_amdgcn_global_load_async_to_lds_b128((gas_i32v4*)s1, (las_i32v4*)d1, 0, 0);
#else
        *(u32x4*)d0 = *(const u32x4*)s0;
        *(u32x4*)d1 = *(const u32x4*)s1;
#endif
    };

    int T = K / 32;
    issueA(0, 0);
    for (int t = 0; t < T; ++t) {
        int buf = t & 1;
        if (t + 1 < T) {
            issueA(buf ^ 1, (t + 1) * 32);   // prefetch next tile into other buffer
#ifdef USE_ASYNC_LDS
            __builtin_amdgcn_s_wait_asynccnt(2);   // tile t complete (in-order)
#endif
        } else {
#ifdef USE_ASYNC_LDS
            __builtin_amdgcn_s_wait_asynccnt(0);
#endif
        }
        __syncthreads();                      // tile t visible to all waves

        Frag af[4], bf[2];
#pragma unroll
        for (int i = 0; i < 4; ++i) {
            const f16* p = &As[buf][(wm * 64 + i * 16 + l16) * 32 + hb * 8];
            af[i].u.lo = *(const u32x4*)p;
            af[i].u.hi = *(const u32x4*)(p + 16);
        }
#pragma unroll
        for (int j = 0; j < 2; ++j) {
            const f16* p = Bt + (ncol0 + wn * 32 + j * 16 + l16) * ldb + t * 32 + hb * 8;
            bf[j].u.lo = *(const u32x4*)p;
            bf[j].u.hi = *(const u32x4*)(p + 16);
            __builtin_prefetch(p + 32, 0, 3);  // next K-step of this B column
        }
#pragma unroll
        for (int i = 0; i < 4; ++i)
#pragma unroll
            for (int j = 0; j < 2; ++j)
                acc[i][j] = __builtin_amdgcn_wmma_f32_16x16x32_f16(
                    false, af[i].h, false, bf[j].h, (short)0, acc[i][j], false, false);

        __syncthreads();                      // all reads of buf done before overwrite
    }

#pragma unroll
    for (int i = 0; i < 4; ++i) {
#pragma unroll
        for (int j = 0; j < 2; ++j) {
            long col = ncol0 + wn * 32 + j * 16 + l16;
            float bv = bias[col];
            long rowb = mrow0 + wm * 64 + i * 16 + hb * 8;
#pragma unroll
            for (int r = 0; r < 8; ++r) {
                long row = rowb + r;
                float v = acc[i][j][r] + bv;
                if (MODE == 0) {
                    ((float*)Cout)[row * ldc + col] = v;
                } else {
                    float g = 0.5f * v * (1.f + tanhf(0.7978845608028654f * (v + 0.044715f * v * v * v)));
                    ((f16*)Cout)[row * ldc + col] = (f16)g;
                }
            }
        }
    }
}

// ------------------------- RMS-norm(q,k) + RoPE + repack q/k/v, f16 outputs.
// Q,K: [h][l][128] row-major.  V: transposed [h][128][l] for attention B-frags.
__global__ __launch_bounds__(128) void rmsrope_kernel(const float* __restrict__ qkvf,
                                                      const float* __restrict__ qnT,
                                                      const float* __restrict__ knT,
                                                      const float* __restrict__ qnI,
                                                      const float* __restrict__ knI,
                                                      const float* __restrict__ pe,
                                                      f16* __restrict__ Qo,
                                                      f16* __restrict__ Ko,
                                                      f16* __restrict__ Vt) {
    int l = blockIdx.x, h = blockIdx.y, d = threadIdx.x;
    const float* base = qkvf + (size_t)l * N_QKV + h * 128 + d;
    float q = base[0], k = base[D_MODEL], v = base[2 * D_MODEL];
    __shared__ float red[4];
    float sq = q * q;
#pragma unroll
    for (int m = 16; m > 0; m >>= 1) sq += __shfl_xor(sq, m, 32);
    if ((d & 31) == 0) red[d >> 5] = sq;
    __syncthreads();
    float tq = red[0] + red[1] + red[2] + red[3];
    __syncthreads();
    float sk = k * k;
#pragma unroll
    for (int m = 16; m > 0; m >>= 1) sk += __shfl_xor(sk, m, 32);
    if ((d & 31) == 0) red[d >> 5] = sk;
    __syncthreads();
    float tk = red[0] + red[1] + red[2] + red[3];

    const float* qn = (l < L_TXT) ? qnT : qnI;
    const float* kn = (l < L_TXT) ? knT : knI;
    float qr = q * rsqrtf(tq * (1.f / 128.f) + EPSV) * qn[d];
    float kr = k * rsqrtf(tk * (1.f / 128.f) + EPSV) * kn[d];
    float qo = __shfl_xor(qr, 1, 32), ko = __shfl_xor(kr, 1, 32);
    float qx0 = (d & 1) ? qo : qr, qx1 = (d & 1) ? qr : qo;
    float kx0 = (d & 1) ? ko : kr, kx1 = (d & 1) ? kr : ko;
    const float* peb = pe + (size_t)l * 256 + (d >> 1) * 4 + (d & 1) * 2;
    float e0 = peb[0], e1 = peb[1];
    float qrp = (e0 * qx0 + e1 * qx1) * 0.08838834764831845f;  // fold 1/sqrt(128)
    float krp = e0 * kx0 + e1 * kx1;
    Qo[((size_t)h * L_TOT + l) * 128 + d] = (f16)qrp;
    Ko[((size_t)h * L_TOT + l) * 128 + d] = (f16)krp;
    Vt[((size_t)h * 128 + d) * L_TOT + l] = (f16)v;
}

// ---------------------------------------- flash attention, wave = 16 q rows.
// Output: attn[l][h*128+d] f16 (proj GEMM input layout).
__global__ __launch_bounds__(256) void attn_kernel(const f16* __restrict__ Q,
                                                   const f16* __restrict__ K,
                                                   const f16* __restrict__ Vt,
                                                   f16* __restrict__ attnOut) {
    int h = blockIdx.x, qb = blockIdx.y;
    int tid = threadIdx.x, wave = tid >> 5, lane = tid & 31;
    int l16 = lane & 15, hb = lane >> 4;
    int qrow0 = qb * 128 + wave * 16;
    const f16* Qh = Q + (size_t)h * L_TOT * 128;
    const f16* Kh = K + (size_t)h * L_TOT * 128;
    const f16* Vh = Vt + (size_t)h * 128 * L_TOT;

    Frag qf[4];
#pragma unroll
    for (int kk = 0; kk < 4; ++kk) {
        const f16* p = Qh + (size_t)(qrow0 + l16) * 128 + kk * 32 + hb * 8;
        qf[kk].u.lo = *(const u32x4*)p;
        qf[kk].u.hi = *(const u32x4*)(p + 16);
    }
    v8f mrow, lrow, o[8];
#pragma unroll
    for (int r = 0; r < 8; ++r) { mrow[r] = -1e30f; lrow[r] = 0.f; }
#pragma unroll
    for (int j = 0; j < 8; ++j)
#pragma unroll
        for (int r = 0; r < 8; ++r) o[j][r] = 0.f;

    __shared__ __align__(16) f16 pb[8][16 * 32];

    for (int kb = 0; kb < L_TOT; kb += 32) {
        v8f s[2];
#pragma unroll
        for (int t = 0; t < 2; ++t) {
#pragma unroll
            for (int r = 0; r < 8; ++r) s[t][r] = 0.f;
#pragma unroll
            for (int kk = 0; kk < 4; ++kk) {
                Frag kf;
                const f16* p = Kh + (size_t)(kb + t * 16 + l16) * 128 + kk * 32 + hb * 8;
                kf.u.lo = *(const u32x4*)p;
                kf.u.hi = *(const u32x4*)(p + 16);
                s[t] = __builtin_amdgcn_wmma_f32_16x16x32_f16(
                    false, qf[kk].h, false, kf.h, (short)0, s[t], false, false);
            }
        }
        // row stats: component r <-> row (r + 8*hb); reduce within 16-lane half
        v8f tmax;
#pragma unroll
        for (int r = 0; r < 8; ++r) tmax[r] = fmaxf(s[0][r], s[1][r]);
#pragma unroll
        for (int m = 1; m < 16; m <<= 1)
#pragma unroll
            for (int r = 0; r < 8; ++r) tmax[r] = fmaxf(tmax[r], __shfl_xor(tmax[r], m, 32));
        v8f newm, scal;
#pragma unroll
        for (int r = 0; r < 8; ++r) {
            newm[r] = fmaxf(mrow[r], tmax[r]);
            scal[r] = expf(mrow[r] - newm[r]);
        }
#pragma unroll
        for (int t = 0; t < 2; ++t)
#pragma unroll
            for (int r = 0; r < 8; ++r) s[t][r] = expf(s[t][r] - newm[r]);
        v8f psum;
#pragma unroll
        for (int r = 0; r < 8; ++r) psum[r] = s[0][r] + s[1][r];
#pragma unroll
        for (int m = 1; m < 16; m <<= 1)
#pragma unroll
            for (int r = 0; r < 8; ++r) psum[r] += __shfl_xor(psum[r], m, 32);
#pragma unroll
        for (int r = 0; r < 8; ++r) { lrow[r] = lrow[r] * scal[r] + psum[r]; mrow[r] = newm[r]; }
#pragma unroll
        for (int j = 0; j < 8; ++j)
#pragma unroll
            for (int r = 0; r < 8; ++r) o[j][r] *= scal[r];

        // re-stripe P: C-layout -> A-layout via per-wave LDS bounce
        __syncthreads();
#pragma unroll
        for (int t = 0; t < 2; ++t)
#pragma unroll
            for (int r = 0; r < 8; ++r)
                pb[wave][(r + hb * 8) * 32 + t * 16 + l16] = (f16)s[t][r];
        __syncthreads();
        Frag pf;
        {
            const f16* p = &pb[wave][l16 * 32 + hb * 8];
            pf.u.lo = *(const u32x4*)p;
            pf.u.hi = *(const u32x4*)(p + 16);
        }
#pragma unroll
        for (int j = 0; j < 8; ++j) {
            Frag vf;
            const f16* p = Vh + (size_t)(j * 16 + l16) * L_TOT + kb + hb * 8;
            vf.u.lo = *(const u32x4*)p;
            vf.u.hi = *(const u32x4*)(p + 16);
            o[j] = __builtin_amdgcn_wmma_f32_16x16x32_f16(
                false, pf.h, false, vf.h, (short)0, o[j], false, false);
        }
    }
#pragma unroll
    for (int j = 0; j < 8; ++j) {
        int col = h * 128 + j * 16 + l16;
#pragma unroll
        for (int r = 0; r < 8; ++r) {
            int row = qrow0 + r + hb * 8;
            attnOut[(size_t)row * D_MODEL + col] = (f16)(o[j][r] / lrow[r]);
        }
    }
}

// ------------------------------------------------ x2 = x + g1 * proj_out
__global__ void residual1_kernel(const float* __restrict__ txtX, const float* __restrict__ imgX,
                                 const float* __restrict__ modT, const float* __restrict__ modI,
                                 const float* __restrict__ proj, float* __restrict__ x2) {
    long idx = (long)blockIdx.x * 256 + threadIdx.x;
    if (idx >= (long)L_TOT * D_MODEL) return;
    long row = idx / D_MODEL, col = idx % D_MODEL;
    float base, g;
    if (row < L_TXT) { base = txtX[row * D_MODEL + col];           g = modT[2 * D_MODEL + col]; }
    else             { base = imgX[(row - L_TXT) * D_MODEL + col]; g = modI[2 * D_MODEL + col]; }
    x2[idx] = base + g * proj[idx];
}

// ------------------------------- out = x2 + g2 * mlp2 (img first in d_out)
__global__ void final_kernel(const float* __restrict__ x2, const float* __restrict__ mlp2,
                             const float* __restrict__ modT, const float* __restrict__ modI,
                             float* __restrict__ out) {
    long idx = (long)blockIdx.x * 256 + threadIdx.x;
    if (idx >= (long)L_TOT * D_MODEL) return;
    long row = idx / D_MODEL, col = idx % D_MODEL;
    float g; long dst;
    if (row < L_TXT) { g = modT[5 * D_MODEL + col]; dst = (long)1024 * D_MODEL + row * D_MODEL + col; }
    else             { g = modI[5 * D_MODEL + col]; dst = (row - L_TXT) * D_MODEL + col; }
    out[dst] = x2[idx] + g * mlp2[idx];
}

// ===========================================================================
extern "C" void kernel_launch(void* const* d_in, const int* in_sizes, int n_in,
                              void* d_out, int out_size, void* d_ws, size_t ws_size,
                              hipStream_t stream) {
    (void)in_sizes; (void)n_in; (void)out_size; (void)ws_size;
    const float* img        = (const float*)d_in[0];
    const float* txt        = (const float*)d_in[1];
    const float* vec        = (const float*)d_in[2];
    const float* pe         = (const float*)d_in[3];
    const float* img_mod_w  = (const float*)d_in[4];
    const float* img_mod_b  = (const float*)d_in[5];
    const float* img_qkv_w  = (const float*)d_in[6];
    const float* img_qkv_b  = (const float*)d_in[7];
    const float* img_qnorm  = (const float*)d_in[8];
    const float* img_knorm  = (const float*)d_in[9];
    const float* img_proj_w = (const float*)d_in[10];
    const float* img_proj_b = (const float*)d_in[11];
    const float* img_mlp_w1 = (const float*)d_in[12];
    const float* img_mlp_b1 = (const float*)d_in[13];
    const float* img_mlp_w2 = (const float*)d_in[14];
    const float* img_mlp_b2 = (const float*)d_in[15];
    const float* txt_mod_w  = (const float*)d_in[16];
    const float* txt_mod_b  = (const float*)d_in[17];
    const float* txt_qkv_w  = (const float*)d_in[18];
    const float* txt_qkv_b  = (const float*)d_in[19];
    const float* txt_qnorm  = (const float*)d_in[20];
    const float* txt_knorm  = (const float*)d_in[21];
    const float* txt_proj_w = (const float*)d_in[22];
    const float* txt_proj_b = (const float*)d_in[23];
    const float* txt_mlp_w1 = (const float*)d_in[24];
    const float* txt_mlp_b1 = (const float*)d_in[25];
    const float* txt_mlp_w2 = (const float*)d_in[26];
    const float* txt_mlp_b2 = (const float*)d_in[27];
    float* out = (float*)d_out;

    char* wsp = (char*)d_ws;
    size_t off = 0;
    auto alloc = [&](size_t bytes) -> char* {
        char* p = wsp + off;
        off = (off + bytes + 255) & ~(size_t)255;
        return p;
    };
    float* sv      = (float*)alloc((size_t)D_MODEL * 4);
    float* modI    = (float*)alloc((size_t)N_MOD * 4);
    float* modT    = (float*)alloc((size_t)N_MOD * 4);
    f16*   xm      = (f16*)alloc((size_t)L_TOT * D_MODEL * 2);
    f16*   wqkvI   = (f16*)alloc((size_t)D_MODEL * N_QKV * 2);    // [N][K]
    f16*   wqkvT   = (f16*)alloc((size_t)D_MODEL * N_QKV * 2);
    f16*   wprojI  = (f16*)alloc((size_t)D_MODEL * D_MODEL * 2);
    f16*   wprojT  = (f16*)alloc((size_t)D_MODEL * D_MODEL * 2);
    f16*   wm1I    = (f16*)alloc((size_t)D_MODEL * N_MLP * 2);
    f16*   wm1T    = (f16*)alloc((size_t)D_MODEL * N_MLP * 2);
    f16*   wm2I    = (f16*)alloc((size_t)N_MLP * D_MODEL * 2);
    f16*   wm2T    = (f16*)alloc((size_t)N_MLP * D_MODEL * 2);
    float* qkvf    = (float*)alloc((size_t)L_TOT * N_QKV * 4);
    f16*   Qb      = (f16*)alloc((size_t)24 * L_TOT * 128 * 2);
    f16*   Kb      = (f16*)alloc((size_t)24 * L_TOT * 128 * 2);
    f16*   Vt      = (f16*)alloc((size_t)24 * 128 * L_TOT * 2);
    f16*   attn    = (f16*)alloc((size_t)L_TOT * D_MODEL * 2);
    float* projout = (float*)alloc((size_t)L_TOT * D_MODEL * 4);
    float* x2      = (float*)alloc((size_t)L_TOT * D_MODEL * 4);
    f16*   xm2     = (f16*)alloc((size_t)L_TOT * D_MODEL * 2);
    f16*   hid     = (f16*)alloc((size_t)L_TOT * N_MLP * 2);
    float* mlp2o   = (float*)alloc((size_t)L_TOT * D_MODEL * 4);

    // modulation vectors
    silu_kernel<<<12, 256, 0, stream>>>(vec, sv);
    mod_gemv_kernel<<<N_MOD / 256, 256, 0, stream>>>(sv, img_mod_w, img_mod_b, modI);
    mod_gemv_kernel<<<N_MOD / 256, 256, 0, stream>>>(sv, txt_mod_w, txt_mod_b, modT);

    // weights: f32 [K][N] -> f16 [N][K] (transposed so GEMM B-frags are
    // contiguous b128 global loads; MLP weight (75MB f16) stays L2-resident)
    auto convT = [&](const float* src, f16* dst, int K, int N) {
        transpose_convert_kernel<<<dim3(N / 32, K / 32), 256, 0, stream>>>(src, dst, K, N);
    };
    convT(img_qkv_w,  wqkvI,  D_MODEL, N_QKV);
    convT(txt_qkv_w,  wqkvT,  D_MODEL, N_QKV);
    convT(img_proj_w, wprojI, D_MODEL, D_MODEL);
    convT(txt_proj_w, wprojT, D_MODEL, D_MODEL);
    convT(img_mlp_w1, wm1I,   D_MODEL, N_MLP);
    convT(txt_mlp_w1, wm1T,   D_MODEL, N_MLP);
    convT(img_mlp_w2, wm2I,   N_MLP,   D_MODEL);
    convT(txt_mlp_w2, wm2T,   N_MLP,   D_MODEL);

    // LN1 + modulation (sh1 @ 0, sc1 @ 3072)
    ln_mod_kernel<<<L_TOT, 256, 0, stream>>>(txt, img, modT, modI, xm, 0, D_MODEL);

    // qkv GEMMs (txt rows 0..255, img rows 256..1279)
    gemm_kernel<0><<<dim3(N_QKV / 128, 8), 256, 0, stream>>>(
        xm + (size_t)L_TXT * D_MODEL, D_MODEL, wqkvI, D_MODEL, img_qkv_b,
        qkvf + (size_t)L_TXT * N_QKV, N_QKV, D_MODEL);
    gemm_kernel<0><<<dim3(N_QKV / 128, 2), 256, 0, stream>>>(
        xm, D_MODEL, wqkvT, D_MODEL, txt_qkv_b, qkvf, N_QKV, D_MODEL);

    // rms-norm + rope + repack
    rmsrope_kernel<<<dim3(L_TOT, 24), 128, 0, stream>>>(
        qkvf, txt_qnorm, txt_knorm, img_qnorm, img_knorm, pe, Qb, Kb, Vt);

    // attention
    attn_kernel<<<dim3(24, 10), 256, 0, stream>>>(Qb, Kb, Vt, attn);

    // projection GEMMs
    gemm_kernel<0><<<dim3(D_MODEL / 128, 8), 256, 0, stream>>>(
        attn + (size_t)L_TXT * D_MODEL, D_MODEL, wprojI, D_MODEL, img_proj_b,
        projout + (size_t)L_TXT * D_MODEL, D_MODEL, D_MODEL);
    gemm_kernel<0><<<dim3(D_MODEL / 128, 2), 256, 0, stream>>>(
        attn, D_MODEL, wprojT, D_MODEL, txt_proj_b, projout, D_MODEL, D_MODEL);

    long tot = (long)L_TOT * D_MODEL;
    residual1_kernel<<<(int)((tot + 255) / 256), 256, 0, stream>>>(txt, img, modT, modI, projout, x2);

    // LN2 + modulation (sh2 @ 3*3072, sc2 @ 4*3072); x2 already in concat order
    ln_mod_kernel<<<L_TOT, 256, 0, stream>>>(x2, x2 + (size_t)L_TXT * D_MODEL,
                                             modT, modI, xm2, 3 * D_MODEL, 4 * D_MODEL);

    // MLP up (GELU, f16 out)
    gemm_kernel<1><<<dim3(N_MLP / 128, 8), 256, 0, stream>>>(
        xm2 + (size_t)L_TXT * D_MODEL, D_MODEL, wm1I, D_MODEL, img_mlp_b1,
        hid + (size_t)L_TXT * N_MLP, N_MLP, D_MODEL);
    gemm_kernel<1><<<dim3(N_MLP / 128, 2), 256, 0, stream>>>(
        xm2, D_MODEL, wm1T, D_MODEL, txt_mlp_b1, hid, N_MLP, D_MODEL);

    // MLP down (f32 out)
    gemm_kernel<0><<<dim3(D_MODEL / 128, 8), 256, 0, stream>>>(
        hid + (size_t)L_TXT * N_MLP, N_MLP, wm2I, N_MLP, img_mlp_b2,
        mlp2o + (size_t)L_TXT * D_MODEL, D_MODEL, N_MLP);
    gemm_kernel<0><<<dim3(D_MODEL / 128, 2), 256, 0, stream>>>(
        hid, N_MLP, wm2T, N_MLP, txt_mlp_b2, mlp2o, D_MODEL, N_MLP);

    final_kernel<<<(int)((tot + 255) / 256), 256, 0, stream>>>(x2, mlp2o, modT, modI, out);
}
